// MultiHeadAttention_75986561401489
// MI455X (gfx1250) — compile-verified, matrix operand order checked by reading, not run on hardware
//
#include <hip/hip_runtime.h>
#include <hip/hip_bf16.h>

// MI455X / gfx1250 wave32 WMMA causal MHA.
// bf16 storage with K-contiguous layouts -> every fragment = 2x b128 loads.
// Pipeline: cvt_x, cvt_wqkv, cvt_wo -> qkv_proj -> flash_attn -> out_proj.

typedef __attribute__((ext_vector_type(16))) __bf16 v16bf;
typedef __attribute__((ext_vector_type(8)))  __bf16 v8bf;   // 16 bytes
typedef __attribute__((ext_vector_type(8)))  float  v8f;
typedef __attribute__((ext_vector_type(4)))  float  v4f;

constexpr int B = 4, T = 2048, E = 1024, H = 16, DH = 64;
constexpr int BT = B * T;                       // 8192
constexpr size_t QKVN = (size_t)B * H * T * DH; // 8,388,608

static __device__ __forceinline__ __bf16 f2bf(float f) { return (__bf16)f; }

static __device__ __forceinline__ v8f wmma_bf16(v16bf a, v16bf b, v8f c) {
  return __builtin_amdgcn_wmma_f32_16x16x32_bf16(false, a, false, b,
                                                 (short)0, c, false, false);
}

// Generic WMMA operand loader; operand's K axis must be contiguous in memory.
// Per ISA 7.12.2: lane owns two contiguous 8-elem runs at k = half*8 (+16).
static __device__ __forceinline__ v16bf load_frag(const __bf16* base, int row0,
                                                  int k0, int ld) {
  int lane = threadIdx.x & 31;
  const __bf16* p =
      base + (size_t)(row0 + (lane & 15)) * ld + k0 + ((lane >> 4) << 3);
  union { v16bf f; v8bf h[2]; } u;
  u.h[0] = *(const v8bf*)p;
  u.h[1] = *(const v8bf*)(p + 16);
  return u.f;
}

// ---- prep kernels ---------------------------------------------------------
__global__ void cvt_x_kernel(const float* __restrict__ x,
                             __bf16* __restrict__ xbf) {
  size_t i = ((size_t)blockIdx.x * blockDim.x + threadIdx.x) * 8;
  v4f a = *(const v4f*)(x + i);
  v4f b = *(const v4f*)(x + i + 4);
  v8bf o;
#pragma unroll
  for (int j = 0; j < 4; ++j) { o[j] = f2bf(a[j]); o[4 + j] = f2bf(b[j]); }
  *(v8bf*)(xbf + i) = o;
}

// Wq/Wk/Wv [H,E,DH] f32 -> wT [3,H,DH,E] bf16 (K=E contiguous).
__global__ void cvt_wqkv_kernel(const float* __restrict__ Wq,
                                const float* __restrict__ Wk,
                                const float* __restrict__ Wv,
                                __bf16* __restrict__ wT) {
  size_t idx = (size_t)blockIdx.x * blockDim.x + threadIdx.x;
  int e = (int)(idx % E);
  int d = (int)((idx / E) % DH);
  int h = (int)((idx / ((size_t)E * DH)) % H);
  int w = (int)(idx / ((size_t)E * DH * H));
  const float* W = (w == 0 ? Wq : w == 1 ? Wk : Wv);
  wT[idx] = f2bf(W[((size_t)h * E + e) * DH + d]);
}

// Wo [E,E] f32 -> woT [E,E] bf16 transposed (woT[n][k] = Wo[k][n]).
__global__ void cvt_wo_kernel(const float* __restrict__ Wo,
                              __bf16* __restrict__ woT) {
  size_t idx = (size_t)blockIdx.x * blockDim.x + threadIdx.x;
  int kidx = (int)(idx % E);
  int nn = (int)(idx / E);
  woT[idx] = f2bf(Wo[(size_t)kidx * E + nn]);
}

// ---- kernel 1: per-head QKV projection ------------------------------------
// grid (BT/128, H, 3), block 128 (4 waves). Wave: 32x64 tile of q|k|v.
// q,k stored [B,H,T,DH]; v stored transposed [B,H,DH,T].
__global__ void qkv_proj_kernel(const __bf16* __restrict__ xbf,
                                const __bf16* __restrict__ wT,
                                __bf16* __restrict__ q,
                                __bf16* __restrict__ k,
                                __bf16* __restrict__ vT) {
  int w = threadIdx.x >> 5;
  int r0 = (blockIdx.x * 4 + w) * 32;            // row of [BT, E]
  int h = blockIdx.y, which = blockIdx.z;
  int b = r0 / T, t0 = r0 % T;
  const __bf16* whead = wT + (size_t)(which * H + h) * DH * E;

  v8f acc[8] = {{}, {}, {}, {}, {}, {}, {}, {}};
#pragma unroll 2
  for (int kk = 0; kk < E; kk += 32) {
    v16bf a0 = load_frag(xbf, r0,      kk, E);
    v16bf a1 = load_frag(xbf, r0 + 16, kk, E);
    v16bf b0 = load_frag(whead, 0,  kk, E);
    v16bf b1 = load_frag(whead, 16, kk, E);
    v16bf b2 = load_frag(whead, 32, kk, E);
    v16bf b3 = load_frag(whead, 48, kk, E);
    acc[0] = wmma_bf16(a0, b0, acc[0]);
    acc[1] = wmma_bf16(a0, b1, acc[1]);
    acc[2] = wmma_bf16(a0, b2, acc[2]);
    acc[3] = wmma_bf16(a0, b3, acc[3]);
    acc[4] = wmma_bf16(a1, b0, acc[4]);
    acc[5] = wmma_bf16(a1, b1, acc[5]);
    acc[6] = wmma_bf16(a1, b2, acc[6]);
    acc[7] = wmma_bf16(a1, b3, acc[7]);
  }

  int lane = threadIdx.x & 31, half = lane >> 4, n = lane & 15;
  if (which == 2) {
    // transposed store: vt[d][t], C-frag rows contiguous in t -> packed v8bf
    __bf16* vt = vT + ((size_t)b * H + h) * DH * T;
#pragma unroll
    for (int mt = 0; mt < 2; ++mt)
#pragma unroll
      for (int dt = 0; dt < 4; ++dt) {
        v8bf pk;
#pragma unroll
        for (int r = 0; r < 8; ++r) pk[r] = f2bf(acc[mt * 4 + dt][r]);
        *(v8bf*)(vt + (size_t)(dt * 16 + n) * T + t0 + mt * 16 + half * 8) = pk;
      }
  } else {
    __bf16* dst = (which == 0 ? q : k) + (((size_t)b * H + h) * T + t0) * DH;
#pragma unroll
    for (int mt = 0; mt < 2; ++mt)
#pragma unroll
      for (int dt = 0; dt < 4; ++dt)
#pragma unroll
        for (int r = 0; r < 8; ++r)
          dst[(size_t)(mt * 16 + r + 8 * half) * DH + dt * 16 + n] =
              f2bf(acc[mt * 4 + dt][r]);
  }
}

// ---- kernel 2: causal flash attention -------------------------------------
// grid (T/16, H, B), block 32 (one wave per 16-query tile).
__global__ void flash_attn_kernel(const __bf16* __restrict__ q,
                                  const __bf16* __restrict__ k,
                                  const __bf16* __restrict__ vT,
                                  __bf16* __restrict__ attn) {
  __shared__ __align__(16) __bf16 ldsP[16 * 32];

  int qt = blockIdx.x, h = blockIdx.y, b = blockIdx.z;
  int q0 = qt * 16;
  const __bf16* qh = q + (((size_t)b * H + h) * T) * DH;
  const __bf16* kh = k + (((size_t)b * H + h) * T) * DH;
  const __bf16* vt = vT + ((size_t)b * H + h) * DH * T;

  int lane = threadIdx.x & 31, half = lane >> 4, n = lane & 15;
  const float scale = 0.125f;  // DH^-0.5

  v16bf qa0 = load_frag(qh, q0, 0, DH);
  v16bf qa1 = load_frag(qh, q0, 32, DH);

  v8f o[4] = {{}, {}, {}, {}};
  float mrow[8], lrow[8];
#pragma unroll
  for (int r = 0; r < 8; ++r) { mrow[r] = -1e30f; lrow[r] = 0.0f; }

  int kbmax = (q0 + 15) / 32;  // inclusive; 32-wide key blocks
  for (int kb = 0; kb <= kbmax; ++kb) {
    int sbase = kb * 32;

    // prefetch next key block's K and V rows into cache (global_prefetch_b8)
    if (kb < kbmax) {
      int snext = sbase + 32;
      __builtin_prefetch(kh + (size_t)(snext + lane) * DH, 0, 3);
      __builtin_prefetch(vt + (size_t)lane * T + snext, 0, 3);
      __builtin_prefetch(vt + (size_t)(32 + lane) * T + snext, 0, 3);
    }

    // hoist all K and V fragment loads: V latency hidden under softmax VALU
    v16bf kb00 = load_frag(kh, sbase,      0,  DH);
    v16bf kb01 = load_frag(kh, sbase,      32, DH);
    v16bf kb10 = load_frag(kh, sbase + 16, 0,  DH);
    v16bf kb11 = load_frag(kh, sbase + 16, 32, DH);
    v16bf pv0 = load_frag(vt, 0,  sbase, T);
    v16bf pv1 = load_frag(vt, 16, sbase, T);
    v16bf pv2 = load_frag(vt, 32, sbase, T);
    v16bf pv3 = load_frag(vt, 48, sbase, T);

    v8f sc[2];
    {
      v8f c = {};
      c = wmma_bf16(qa0, kb00, c);
      sc[0] = wmma_bf16(qa1, kb01, c);
      v8f d = {};
      d = wmma_bf16(qa0, kb10, d);
      sc[1] = wmma_bf16(qa1, kb11, d);
    }
    // scale + causal mask: keep element (m,n) iff s <= q
#pragma unroll
    for (int r = 0; r < 8; ++r) {
      int qrow = q0 + r + 8 * half;
      sc[0][r] = (sbase + n      <= qrow) ? sc[0][r] * scale : -1e30f;
      sc[1][r] = (sbase + 16 + n <= qrow) ? sc[1][r] * scale : -1e30f;
    }
    // online softmax (each lane owns 8 rows across the C fragments)
#pragma unroll
    for (int r = 0; r < 8; ++r) {
      float mx = fmaxf(sc[0][r], sc[1][r]);
#pragma unroll
      for (int off = 8; off; off >>= 1) mx = fmaxf(mx, __shfl_xor(mx, off, 32));
      float mnew  = fmaxf(mrow[r], mx);
      float alpha = __expf(mrow[r] - mnew);
      float p0 = __expf(sc[0][r] - mnew);
      float p1 = __expf(sc[1][r] - mnew);
      float rs = p0 + p1;
#pragma unroll
      for (int off = 8; off; off >>= 1) rs += __shfl_xor(rs, off, 32);
      lrow[r] = lrow[r] * alpha + rs;
      mrow[r] = mnew;
      sc[0][r] = p0; sc[1][r] = p1;
#pragma unroll
      for (int dt = 0; dt < 4; ++dt) o[dt][r] *= alpha;
    }
    // stage P (16x32) to LDS: C-layout -> row-major for the A fragment
#pragma unroll
    for (int r = 0; r < 8; ++r) {
      int m = r + 8 * half;
      ldsP[m * 32 + n]      = f2bf(sc[0][r]);
      ldsP[m * 32 + 16 + n] = f2bf(sc[1][r]);
    }
    __syncthreads();
    asm volatile("s_wait_dscnt 0x0" ::: "memory");  // wave-local LDS RAW fence

    v16bf pa = load_frag(ldsP, 0, 0, 32);
    o[0] = wmma_bf16(pa, pv0, o[0]);
    o[1] = wmma_bf16(pa, pv1, o[1]);
    o[2] = wmma_bf16(pa, pv2, o[2]);
    o[3] = wmma_bf16(pa, pv3, o[3]);
    __syncthreads();
  }

  // normalize, store concat-head layout [B, T, H*DH] bf16
#pragma unroll
  for (int r = 0; r < 8; ++r) {
    int t = q0 + r + 8 * half;
    float inv = 1.0f / lrow[r];
    __bf16* dst = attn + ((size_t)b * T + t) * E + h * DH;
#pragma unroll
    for (int dt = 0; dt < 4; ++dt)
      dst[dt * 16 + n] = f2bf(o[dt][r] * inv);
  }
}

// ---- kernel 3: output projection + bias -----------------------------------
// grid (BT/128, E/64), block 128 (4 waves). Wave: 32x64 tile of out (f32).
__global__ void out_proj_kernel(const __bf16* __restrict__ attn,
                                const __bf16* __restrict__ woT,
                                const float* __restrict__ bo,
                                float* __restrict__ out) {
  int w = threadIdx.x >> 5;
  int r0 = (blockIdx.x * 4 + w) * 32;
  int col0 = blockIdx.y * 64;

  v8f acc[8] = {{}, {}, {}, {}, {}, {}, {}, {}};
#pragma unroll 2
  for (int kk = 0; kk < E; kk += 32) {
    v16bf a0 = load_frag(attn, r0,      kk, E);
    v16bf a1 = load_frag(attn, r0 + 16, kk, E);
    v16bf b0 = load_frag(woT, col0,      kk, E);
    v16bf b1 = load_frag(woT, col0 + 16, kk, E);
    v16bf b2 = load_frag(woT, col0 + 32, kk, E);
    v16bf b3 = load_frag(woT, col0 + 48, kk, E);
    acc[0] = wmma_bf16(a0, b0, acc[0]);
    acc[1] = wmma_bf16(a0, b1, acc[1]);
    acc[2] = wmma_bf16(a0, b2, acc[2]);
    acc[3] = wmma_bf16(a0, b3, acc[3]);
    acc[4] = wmma_bf16(a1, b0, acc[4]);
    acc[5] = wmma_bf16(a1, b1, acc[5]);
    acc[6] = wmma_bf16(a1, b2, acc[6]);
    acc[7] = wmma_bf16(a1, b3, acc[7]);
  }
  int lane = threadIdx.x & 31, half = lane >> 4, n = lane & 15;
#pragma unroll
  for (int dt = 0; dt < 4; ++dt) {
    float bias = bo[col0 + dt * 16 + n];
#pragma unroll
    for (int mt = 0; mt < 2; ++mt)
#pragma unroll
      for (int r = 0; r < 8; ++r)
        out[(size_t)(r0 + mt * 16 + r + 8 * half) * E + col0 + dt * 16 + n] =
            acc[mt * 4 + dt][r] + bias;
  }
}

extern "C" void kernel_launch(void* const* d_in, const int* in_sizes, int n_in,
                              void* d_out, int out_size, void* d_ws, size_t ws_size,
                              hipStream_t stream) {
  const float* x  = (const float*)d_in[0];
  const float* Wq = (const float*)d_in[1];
  const float* Wk = (const float*)d_in[2];
  const float* Wv = (const float*)d_in[3];
  const float* Wo = (const float*)d_in[4];
  const float* bo = (const float*)d_in[5];
  float* out = (float*)d_out;

  // ws layout (bf16 elems): xbf | wT | woT | q | k | vT | attn   (~88 MB)
  __bf16* xbf  = (__bf16*)d_ws;
  __bf16* wT   = xbf  + (size_t)BT * E;
  __bf16* woT  = wT   + (size_t)3 * H * DH * E;
  __bf16* qb   = woT  + (size_t)E * E;
  __bf16* kb   = qb   + QKVN;
  __bf16* vT   = kb   + QKVN;
  __bf16* attn = vT   + QKVN;

  cvt_x_kernel   <<<(size_t)BT * E / (256 * 8), 256, 0, stream>>>(x, xbf);
  cvt_wqkv_kernel<<<(size_t)3 * H * DH * E / 256, 256, 0, stream>>>(Wq, Wk, Wv, wT);
  cvt_wo_kernel  <<<(size_t)E * E / 256, 256, 0, stream>>>(Wo, woT);

  qkv_proj_kernel <<<dim3(BT / 128, H, 3), 128, 0, stream>>>(xbf, wT, qb, kb, vT);
  flash_attn_kernel<<<dim3(T / 16, H, B), 32, 0, stream>>>(qb, kb, vT, attn);
  out_proj_kernel <<<dim3(BT / 128, E / 64), 128, 0, stream>>>(attn, woT, bo, out);
}
